// GCLKG_35553739276533
// MI455X (gfx1250) — matrix operand order repaired and checked by reading, not compiled, not run on hardware
//
#include <hip/hip_runtime.h>

#define N_USERS 100000
#define N_ITEMS 50000
#define N_NODES 150000
#define EMB 64
#define N_EDGES 4000000
#define TPAD 72
#define LN_EPS 1e-5f

typedef __attribute__((ext_vector_type(16))) _Float16 v16h;
typedef __attribute__((ext_vector_type(8)))  float    v8f;

// ---------------- WMMA fragment builders (CDNA5 16x16x32 f16 layouts) ----------------

// A: 16x32 f16. lane m = lane&15 owns row m; VGPR pair index v: K = kb + 2*(v&3) + 16*(v>>2) + 8*(lane>>4)
__device__ __forceinline__ v16h frag_a_f32(const float* __restrict__ Abase, int lda,
                                           int row0, int kb, int lane) {
  const int g = lane >> 4, m = lane & 15;
  const float* rp = Abase + (size_t)(row0 + m) * (size_t)lda;
  v16h a;
#pragma unroll
  for (int v = 0; v < 8; ++v) {
    const int k = kb + 2 * (v & 3) + 16 * (v >> 2) + 8 * g;
    a[2 * v]     = (_Float16)rp[k];
    a[2 * v + 1] = (_Float16)rp[k + 1];
  }
  return a;
}

// Attention A: k<64 -> u_mean[k] (broadcast row), k>=64 -> item_emb[row][k-64]
__device__ __forceinline__ v16h frag_a_attn(const float* __restrict__ um,
                                            const float* __restrict__ iemb,
                                            int row0, int kb, int lane) {
  const int g = lane >> 4, m = lane & 15;
  const float* rp = iemb + (size_t)(row0 + m) * EMB;
  v16h a;
#pragma unroll
  for (int v = 0; v < 8; ++v) {
    const int k = kb + 2 * (v & 3) + 16 * (v >> 2) + 8 * g;
    float x0, x1;
    if (k < EMB) { x0 = um[k]; x1 = um[k + 1]; }            // uniform per kb (kb<64)
    else         { x0 = rp[k - EMB]; x1 = rp[k - EMB + 1]; } // uniform per kb (kb>=64)
    a[2 * v]     = (_Float16)x0;
    a[2 * v + 1] = (_Float16)x1;
  }
  return a;
}

// B: 32x16 f16. lane n = nb + (lane&15); half-slot s: K = kb + 8*(lane>>4) + (s&7) + 16*(s>>3). W is [K][64].
__device__ __forceinline__ v16h frag_b_f32(const float* __restrict__ W, int kb, int nb, int lane) {
  const int g = lane >> 4, n = nb + (lane & 15);
  v16h b;
#pragma unroll
  for (int s = 0; s < 16; ++s)
    b[s] = (_Float16)W[(size_t)(kb + 8 * g + (s & 7) + 16 * (s >> 3)) * EMB + n];
  return b;
}

// 16 rows x 64 cols GEMM: D += A[16xK] * W[Kx64], f16 inputs / f32 accum
__device__ __forceinline__ void wave_gemm(const float* __restrict__ A, int lda, int row0,
                                          const float* __restrict__ W, int K, int lane, v8f c[4]) {
  for (int kb = 0; kb < K; kb += 32) {
    v16h a = frag_a_f32(A, lda, row0, kb, lane);
#pragma unroll
    for (int f = 0; f < 4; ++f) {
      v16h b = frag_b_f32(W, kb, f * 16, lane);
      c[f] = __builtin_amdgcn_wmma_f32_16x16x32_f16(false, a, false, b, (short)0, c[f],
                                                    false, false);
    }
  }
}

// D layout: frag f, VGPR d, lane L -> (M = d + 8*(L>>4), N = f*16 + (L&15))
__device__ __forceinline__ void store_tile_bias(float* __restrict__ tile, const v8f c[4],
                                                const float* __restrict__ bias, int lane) {
  const int g = lane >> 4, n0 = lane & 15;
#pragma unroll
  for (int f = 0; f < 4; ++f) {
    const float bv = bias[f * 16 + n0];
#pragma unroll
    for (int d = 0; d < 8; ++d)
      tile[(d + 8 * g) * TPAD + f * 16 + n0] = c[f][d] + bv;
  }
}

// LayerNorm + LeakyReLU(0.2) on one 16x64 tile; call with lane<16, row = lane
__device__ __forceinline__ void ln_leaky(float* __restrict__ tile,
                                         const float* __restrict__ gam,
                                         const float* __restrict__ bet, int r) {
  float* row = tile + r * TPAD;
  float s = 0.f, ss = 0.f;
#pragma unroll
  for (int c = 0; c < EMB; ++c) { float x = row[c]; s += x; ss += x * x; }
  const float mu   = s * (1.f / EMB);
  const float var  = ss * (1.f / EMB) - mu * mu;
  const float rstd = rsqrtf(var + LN_EPS);
#pragma unroll
  for (int c = 0; c < EMB; ++c) {
    float y = (row[c] - mu) * rstd * gam[c] + bet[c];
    row[c] = (y < 0.f) ? 0.2f * y : y;
  }
}

// ---------------- small utility kernels ----------------

__global__ void __launch_bounds__(256) k_fill_zero(float* __restrict__ p, size_t n) {
  for (size_t i = blockIdx.x * (size_t)blockDim.x + threadIdx.x; i < n;
       i += (size_t)gridDim.x * blockDim.x)
    p[i] = 0.f;
}

__global__ void __launch_bounds__(256) k_copy(float* __restrict__ dst,
                                              const float* __restrict__ src, size_t n) {
  for (size_t i = blockIdx.x * (size_t)blockDim.x + threadIdx.x; i < n;
       i += (size_t)gridDim.x * blockDim.x)
    dst[i] = src[i];
}

// column sums of an [nrows,64] matrix into acc[64]
__global__ void __launch_bounds__(256) k_colsum(const float* __restrict__ x, int nrows,
                                                float* __restrict__ acc) {
  __shared__ float s[64];
  if (threadIdx.x < 64) s[threadIdx.x] = 0.f;
  __syncthreads();
  const size_t n = (size_t)nrows * EMB;
  for (size_t i = blockIdx.x * (size_t)blockDim.x + threadIdx.x; i < n;
       i += (size_t)gridDim.x * blockDim.x)
    atomicAdd(&s[i & 63], x[i]);
  __syncthreads();
  if (threadIdx.x < 64) atomicAdd(&acc[threadIdx.x], s[threadIdx.x]);
}

__global__ void k_small_finalize(const float* __restrict__ u_sum,
                                 const float* __restrict__ m2_sum,
                                 float* __restrict__ u_mean, float* __restrict__ m2_row) {
  const int t = threadIdx.x;
  if (t < 64) {
    u_mean[t] = u_sum[t]  * (1.0f / N_USERS);
    m2_row[t] = m2_sum[t] * (1.0f / N_USERS);
  }
}

// ---------------- encode modal_feat2 and accumulate column sums ----------------

__global__ void __launch_bounds__(256) k_encode_m2(const float* __restrict__ feat,
                                                   const float* __restrict__ W,
                                                   const float* __restrict__ bias,
                                                   const float* __restrict__ gam,
                                                   const float* __restrict__ bet,
                                                   float* __restrict__ m2_sum) {
  __shared__ float tiles[8][16 * TPAD];
  const int lane  = threadIdx.x & 31;
  const int wslot = threadIdx.x >> 5;
  const int gw    = blockIdx.x * 8 + wslot;
  const int ntiles = N_USERS / 16;
  const bool active = gw < ntiles;
  const int row0 = (active ? gw : 0) * 16;
  float* tile = tiles[wslot];

  v8f c[4] = {};
  wave_gemm(feat, 128, row0, W, 128, lane, c);
  store_tile_bias(tile, c, bias, lane);
  __syncthreads();
  if (lane < 16) ln_leaky(tile, gam, bet, lane);
  __syncthreads();
  if (active) {
    const int c0 = lane * 2;
    float s0 = 0.f, s1 = 0.f;
#pragma unroll
    for (int r = 0; r < 16; ++r) {
      s0 += tile[r * TPAD + c0];
      s1 += tile[r * TPAD + c0 + 1];
    }
    atomicAdd(&m2_sum[c0], s0);
    atomicAdd(&m2_sum[c0 + 1], s1);
  }
}

// ---------------- fused item kernel: attention + m0 + m1 + combine ----------------

__global__ void __launch_bounds__(256) k_items(
    const float* __restrict__ item_emb,
    const float* __restrict__ mf0, const float* __restrict__ mf1,
    const float* __restrict__ W0, const float* __restrict__ b0,
    const float* __restrict__ g0, const float* __restrict__ be0,
    const float* __restrict__ W1, const float* __restrict__ b1,
    const float* __restrict__ g1, const float* __restrict__ be1,
    const float* __restrict__ aW1, const float* __restrict__ ab1,
    const float* __restrict__ aW2, const float* __restrict__ ab2,
    const float* __restrict__ u_mean, const float* __restrict__ m2_row,
    float* __restrict__ all_emb) {
  __shared__ float tiles[8][16 * TPAD];
  __shared__ float wlds[8][16][4];
  const int lane  = threadIdx.x & 31;
  const int wslot = threadIdx.x >> 5;
  const int gw    = blockIdx.x * 8 + wslot;
  const int ntiles = N_ITEMS / 16;
  const bool active = gw < ntiles;
  const int row0 = (active ? gw : 0) * 16;
  float* tile = tiles[wslot];

  // --- attention hidden: h = [u_mean | item_emb] @ aW1 + ab1  (16x128 * 128x64) ---
  {
    v8f c[4] = {};
    for (int kb = 0; kb < 2 * EMB; kb += 32) {
      v16h a = frag_a_attn(u_mean, item_emb, row0, kb, lane);
#pragma unroll
      for (int f = 0; f < 4; ++f) {
        v16h b = frag_b_f32(aW1, kb, f * 16, lane);
        c[f] = __builtin_amdgcn_wmma_f32_16x16x32_f16(false, a, false, b, (short)0, c[f],
                                                      false, false);
      }
    }
    store_tile_bias(tile, c, ab1, lane);
  }
  __syncthreads();
  // --- logits + softmax over 3 modalities (lanes 0..15, one row each) ---
  if (lane < 16) {
    const float* row = tile + lane * TPAD;
    float l0 = ab2[0], l1 = ab2[1], l2 = ab2[2];
#pragma unroll 8
    for (int k = 0; k < EMB; ++k) {
      const float h = tanhf(row[k]);
      l0 += h * aW2[k * 3 + 0];
      l1 += h * aW2[k * 3 + 1];
      l2 += h * aW2[k * 3 + 2];
    }
    const float mx = fmaxf(l0, fmaxf(l1, l2));
    const float e0 = __expf(l0 - mx), e1 = __expf(l1 - mx), e2 = __expf(l2 - mx);
    const float inv = 1.f / (e0 + e1 + e2);
    wlds[wslot][lane][0] = e0 * inv;
    wlds[wslot][lane][1] = e1 * inv;
    wlds[wslot][lane][2] = e2 * inv;
  }
  __syncthreads();

  // --- per-lane output accumulator: 16 rows x 2 cols; init item_emb + w2*m2_row ---
  const int c0 = lane * 2;
  float accx[16], accy[16];
#pragma unroll
  for (int r = 0; r < 16; ++r) {
    const float w2 = wlds[wslot][r][2];
    accx[r] = item_emb[(size_t)(row0 + r) * EMB + c0]     + w2 * m2_row[c0];
    accy[r] = item_emb[(size_t)(row0 + r) * EMB + c0 + 1] + w2 * m2_row[c0 + 1];
  }

  // --- modality 0: encode(modal_feat0 @ W0) ---
  {
    v8f c[4] = {};
    wave_gemm(mf0, 768, row0, W0, 768, lane, c);
    __syncthreads();
    store_tile_bias(tile, c, b0, lane);
  }
  __syncthreads();
  if (lane < 16) ln_leaky(tile, g0, be0, lane);
  __syncthreads();
#pragma unroll
  for (int r = 0; r < 16; ++r) {
    const float w0 = wlds[wslot][r][0];
    accx[r] += w0 * tile[r * TPAD + c0];
    accy[r] += w0 * tile[r * TPAD + c0 + 1];
  }
  __syncthreads();

  // --- modality 1: encode(modal_feat1 @ W1) ---
  {
    v8f c[4] = {};
    wave_gemm(mf1, 384, row0, W1, 384, lane, c);
    store_tile_bias(tile, c, b1, lane);
  }
  __syncthreads();
  if (lane < 16) ln_leaky(tile, g1, be1, lane);
  __syncthreads();
#pragma unroll
  for (int r = 0; r < 16; ++r) {
    const float w1 = wlds[wslot][r][1];
    accx[r] += w1 * tile[r * TPAD + c0];
    accy[r] += w1 * tile[r * TPAD + c0 + 1];
  }

  if (active) {
#pragma unroll
    for (int r = 0; r < 16; ++r) {
      float2 v = make_float2(accx[r], accy[r]);
      *(float2*)(all_emb + (size_t)(N_USERS + row0 + r) * EMB + c0) = v;
    }
  }
}

// ---------------- COO SpMM: eout[row] += val * ein[col], one wave per edge ----------------

__global__ void __launch_bounds__(256) k_spmm(const float* __restrict__ vals,
                                              const int* __restrict__ rows,
                                              const int* __restrict__ cols,
                                              const float* __restrict__ ein,
                                              float* __restrict__ eout) {
  const int lane = threadIdx.x & 31;  // wave32
  const int wid    = (blockIdx.x * blockDim.x + threadIdx.x) >> 5;
  const int nwaves = (gridDim.x * blockDim.x) >> 5;
  const int c0 = lane * 2;
  for (int e = wid; e < N_EDGES; e += nwaves) {
    const float v = vals[e];
    const int r = rows[e], c = cols[e];
    const float2 x = *(const float2*)(ein + (size_t)c * EMB + c0);
    atomicAdd(&eout[(size_t)r * EMB + c0],     v * x.x);
    atomicAdd(&eout[(size_t)r * EMB + c0 + 1], v * x.y);
  }
}

__global__ void __launch_bounds__(256) k_combine(float* __restrict__ out,
                                                 const float* __restrict__ A,
                                                 const float* __restrict__ B, size_t n) {
  for (size_t i = blockIdx.x * (size_t)blockDim.x + threadIdx.x; i < n;
       i += (size_t)gridDim.x * blockDim.x)
    out[i] = (out[i] + A[i] + B[i]) * (1.0f / 3.0f);
}

// ---------------- host launch ----------------

extern "C" void kernel_launch(void* const* d_in, const int* in_sizes, int n_in,
                              void* d_out, int out_size, void* d_ws, size_t ws_size,
                              hipStream_t stream) {
  const float* user_emb = (const float*)d_in[0];
  const float* item_emb = (const float*)d_in[1];
  const float* mf0 = (const float*)d_in[2];
  const float* mf1 = (const float*)d_in[3];
  const float* mf2 = (const float*)d_in[4];
  const float* W0  = (const float*)d_in[5];
  const float* b0  = (const float*)d_in[6];
  const float* g0  = (const float*)d_in[7];
  const float* be0 = (const float*)d_in[8];
  const float* W1  = (const float*)d_in[9];
  const float* b1  = (const float*)d_in[10];
  const float* g1  = (const float*)d_in[11];
  const float* be1 = (const float*)d_in[12];
  const float* W2  = (const float*)d_in[13];
  const float* b2  = (const float*)d_in[14];
  const float* g2  = (const float*)d_in[15];
  const float* be2 = (const float*)d_in[16];
  const float* aW1 = (const float*)d_in[17];
  const float* ab1 = (const float*)d_in[18];
  const float* aW2 = (const float*)d_in[19];
  const float* ab2 = (const float*)d_in[20];
  const float* adj_vals = (const float*)d_in[21];
  const int*   adj_rows = (const int*)d_in[22];
  const int*   adj_cols = (const int*)d_in[23];
  float* out = (float*)d_out;

  const size_t NE = (size_t)N_NODES * EMB;  // 9.6M floats
  float* A      = (float*)d_ws;             // all_emb
  float* B      = A + NE;                   // e1
  float* u_sum  = B + NE;
  float* m2_sum = u_sum + 64;
  float* u_mean = m2_sum + 64;
  float* m2_row = u_mean + 64;

  k_fill_zero<<<2048, 256, 0, stream>>>(B, NE);
  k_fill_zero<<<2048, 256, 0, stream>>>(out, NE);
  k_fill_zero<<<1, 256, 0, stream>>>(u_sum, 128);  // u_sum + m2_sum

  k_colsum<<<1024, 256, 0, stream>>>(user_emb, N_USERS, u_sum);
  k_encode_m2<<<(N_USERS / 16 + 7) / 8, 256, 0, stream>>>(mf2, W2, b2, g2, be2, m2_sum);
  k_small_finalize<<<1, 64, 0, stream>>>(u_sum, m2_sum, u_mean, m2_row);

  k_copy<<<2048, 256, 0, stream>>>(A, user_emb, (size_t)N_USERS * EMB);
  k_items<<<(N_ITEMS / 16 + 7) / 8, 256, 0, stream>>>(
      item_emb, mf0, mf1, W0, b0, g0, be0, W1, b1, g1, be1,
      aW1, ab1, aW2, ab2, u_mean, m2_row, A);

  k_spmm<<<4096, 256, 0, stream>>>(adj_vals, adj_rows, adj_cols, A, B);
  k_spmm<<<4096, 256, 0, stream>>>(adj_vals, adj_rows, adj_cols, B, out);
  k_combine<<<4096, 256, 0, stream>>>(out, A, B, NE);
}
